// L3TAGConv_84859963834425
// MI455X (gfx1250) — compile-verified, matrix operand order checked by reading, not run on hardware
//
#include <hip/hip_runtime.h>

#define DIN  64
#define H1   400
#define H2   200
#define DOUT 4

typedef __attribute__((ext_vector_type(16))) __bf16 v16b;
typedef __attribute__((ext_vector_type(8)))  __bf16 v8b;
typedef __attribute__((ext_vector_type(8)))  float  v8f;

static __device__ __forceinline__ __bf16 f2bf(float f) {
  unsigned u = __builtin_bit_cast(unsigned, f);
  u += 0x7FFFu + ((u >> 16) & 1u);            // round-to-nearest-even
  unsigned short h = (unsigned short)(u >> 16);
  return __builtin_bit_cast(__bf16, h);
}

__global__ void k_zero(float* __restrict__ p, long long n) {
  long long i = (long long)blockIdx.x * blockDim.x + threadIdx.x;
  if (i < n) p[i] = 0.f;
}

__global__ void k_deg(const int* __restrict__ dst, float* __restrict__ deg, int E) {
  int e = blockIdx.x * blockDim.x + threadIdx.x;
  if (e < E)
    __hip_atomic_fetch_add(&deg[dst[e]], 1.f, __ATOMIC_RELAXED, __HIP_MEMORY_SCOPE_AGENT);
}

__global__ void k_dinv(float* __restrict__ deg, int N) {
  int i = blockIdx.x * blockDim.x + threadIdx.x;
  if (i < N) { float d = deg[i]; deg[i] = d > 0.f ? rsqrtf(fmaxf(d, 1.f)) : 0.f; }
}

__global__ void k_norm(const int* __restrict__ src, const int* __restrict__ dst,
                       const float* __restrict__ dinv, float* __restrict__ norm, int E) {
  int e = blockIdx.x * blockDim.x + threadIdx.x;
  if (e < E) norm[e] = dinv[src[e]] * dinv[dst[e]];
}

// out[dst,:] += norm[e] * h[src,:]  (edge-parallel, L2-resident atomics)
__global__ void k_prop(const float* __restrict__ h, int ldh,
                       float* __restrict__ out, int ldo,
                       const float* __restrict__ norm,
                       const int* __restrict__ src, const int* __restrict__ dst,
                       int F, long long total) {
  long long i = (long long)blockIdx.x * blockDim.x + threadIdx.x;
  if (i >= total) return;
  int e = (int)(i / F);
  int f = (int)(i - (long long)e * F);
  float v = norm[e] * h[(size_t)src[e] * ldh + f];
  __hip_atomic_fetch_add(&out[(size_t)dst[e] * ldo + f], v,
                         __ATOMIC_RELAXED, __HIP_MEMORY_SCOPE_AGENT);
}

// f32 -> bf16 staging with zero padding (rows to NP, cols to K-multiple-of-32)
__global__ void k_cvt(const float* __restrict__ s, int ldS, __bf16* __restrict__ d,
                      int ldD, int cols, int rowsReal, long long total) {
  long long i = (long long)blockIdx.x * blockDim.x + threadIdx.x;
  if (i >= total) return;
  long long r = i / ldD; int c = (int)(i - r * ldD);
  float v = (r < rowsReal && c < cols) ? s[r * (long long)ldS + c] : 0.f;
  d[i] = f2bf(v);
}

// Pack W (Kreal x Nreal, row-major f32) into WMMA B-fragment-major bf16:
//   Bp[((kt*Ntiles+nt)*32 + lane)*16 + e] = W[kmap, nt*16 + (lane&15)]
//   kmap = kt*32 + (e<8 ? e : e+8) + (lane>=16 ? 8 : 0)
__global__ void k_pack(const float* __restrict__ W, __bf16* __restrict__ Bp,
                       int Kreal, int Nreal, int Ntiles, int total) {
  int i = blockIdx.x * blockDim.x + threadIdx.x;
  if (i >= total) return;
  int e = i & 15, lane = (i >> 4) & 31, t = i >> 9;
  int nt = t % Ntiles, kt = t / Ntiles;
  int k = kt * 32 + (e < 8 ? e : e + 8) + ((lane >= 16) ? 8 : 0);
  int n = nt * 16 + (lane & 15);
  float v = (k < Kreal && n < Nreal) ? W[(size_t)k * Nreal + n] : 0.f;
  Bp[i] = f2bf(v);
}

// One wave -> one 16x16 C tile.  C (f32) is read-modify-written so GEMM
// accumulates on top of propagated Horner terms.
__global__ void __launch_bounds__(256)
k_gemm(const __bf16* __restrict__ A, int lda,
       const __bf16* __restrict__ Bp, float* __restrict__ C, int ldc,
       int Mtiles, int Ntiles, int Ktiles) {
  int wave = blockIdx.x * (blockDim.x >> 5) + (threadIdx.x >> 5);
  if (wave >= Mtiles * Ntiles) return;
  int lane = threadIdx.x & 31;
  int nt = wave % Ntiles, mt = wave / Ntiles;
  int m0 = mt << 4, n0 = nt << 4;
  int hi = lane >> 4, lm = lane & 15;

  v8f acc;
#pragma unroll
  for (int g = 0; g < 8; ++g)
    acc[g] = C[(size_t)(m0 + g + hi * 8) * ldc + n0 + lm];

  // A fragment: lane holds row (m0+lm); K elems {k0..k0+7, k0+16..k0+23} (+8 for hi lanes)
  const __bf16* arow = A + (size_t)(m0 + lm) * lda + hi * 8;
  const __bf16* bptr = Bp + ((size_t)nt * 32 + lane) * 16;
  size_t bstep = (size_t)Ntiles * 512;

  for (int kt = 0; kt < Ktiles; ++kt) {
    v8b alo = *(const v8b*)(arow);            // 16B aligned
    v8b ahi = *(const v8b*)(arow + 16);
    v16b a = __builtin_shufflevector(alo, ahi, 0,1,2,3,4,5,6,7,8,9,10,11,12,13,14,15);
    v16b b = *(const v16b*)(bptr);            // 32B contiguous per lane
    acc = __builtin_amdgcn_wmma_f32_16x16x32_bf16(false, a, false, b,
                                                  (short)0, acc, false, false);
    arow += 32; bptr += bstep;
  }
#pragma unroll
  for (int g = 0; g < 8; ++g)
    C[(size_t)(m0 + g + hi * 8) * ldc + n0 + lm] = acc[g];
}

__global__ void k_bias_relu(float* __restrict__ o, int ld, const float* __restrict__ b,
                            int cols, long long total) {
  long long i = (long long)blockIdx.x * blockDim.x + threadIdx.x;
  if (i >= total) return;
  long long r = i / cols; int c = (int)(i - r * cols);
  size_t p = (size_t)r * ld + c;
  float v = o[p] + b[c];
  o[p] = v > 0.f ? v : 0.f;
}

__global__ void k_final(const float* __restrict__ o, int ld, const float* __restrict__ b,
                        float* __restrict__ out, int N) {
  int i = blockIdx.x * blockDim.x + threadIdx.x;
  if (i >= N * DOUT) return;
  int r = i >> 2, c = i & 3;
  float v = o[(size_t)r * ld + c] + b[c];
  out[i] = v > 0.f ? v : 0.f;
}

static inline unsigned cdiv(long long a, long long b) { return (unsigned)((a + b - 1) / b); }

extern "C" void kernel_launch(void* const* d_in, const int* in_sizes, int n_in,
                              void* d_out, int out_size, void* d_ws, size_t ws_size,
                              hipStream_t stream) {
  const float* x  = (const float*)d_in[0];
  const int*   ei = (const int*)d_in[1];
  const float* W1 = (const float*)d_in[2];
  const float* b1 = (const float*)d_in[3];
  const float* W2 = (const float*)d_in[4];
  const float* b2 = (const float*)d_in[5];
  const float* W3 = (const float*)d_in[6];
  const float* b3 = (const float*)d_in[7];
  float* out = (float*)d_out;

  const int N  = in_sizes[0] / DIN;
  const int E  = in_sizes[1] / 2;
  const int NP = ((N + 15) / 16) * 16;
  const int Mt = NP / 16;
  const int K1 = 64;    // DIN, already x32
  const int K2 = 416;   // H1=400 padded to x32
  const int K3 = 224;   // H2=200 padded to x32
  const int LD2 = 208;  // H2 padded to x16 (f32 accumulators)
  const int LD3 = 16;   // DOUT padded to x16

  const int* src = ei;
  const int* dst = ei + E;

  char* base = (char*)d_ws;
  size_t off = 0;
  auto alloc = [&](size_t bytes) -> void* {
    void* p = base + off; off = (off + bytes + 255) & ~(size_t)255; return p;
  };
  float*  deg  = (float*) alloc((size_t)N  * 4);         // reused as dinv
  float*  norm = (float*) alloc((size_t)E  * 4);
  float*  hA   = (float*) alloc((size_t)NP * DIN * 4);   // also layer-3 accum (ld 16)
  float*  hB   = (float*) alloc((size_t)NP * DIN * 4);
  float*  out1 = (float*) alloc((size_t)NP * H1  * 4);
  float*  gA   = (float*) alloc((size_t)NP * LD2 * 4);
  float*  gB   = (float*) alloc((size_t)NP * LD2 * 4);
  __bf16* Abf  = (__bf16*)alloc((size_t)NP * K2  * 2);
  __bf16* Wp   = (__bf16*)alloc((size_t)512 * 1024);

  auto Z = [&](float* p, long long n) { k_zero<<<cdiv(n,256),256,0,stream>>>(p, n); };
  auto PROP = [&](const float* hin, int ldh, float* hout, int ldo, int F) {
    long long tot = (long long)E * F;
    k_prop<<<cdiv(tot,256),256,0,stream>>>(hin, ldh, hout, ldo, norm, src, dst, F, tot);
  };
  auto CVT = [&](const float* s, int ldS, int cols, int ldD) {
    long long tot = (long long)NP * ldD;
    k_cvt<<<cdiv(tot,256),256,0,stream>>>(s, ldS, Abf, ldD, cols, N, tot);
  };
  auto GEMM = [&](const float* Wm, int Kreal, int Nreal, int Kt, int Nt, int lda,
                  float* C, int ldc) {
    int ptot = Kt * Nt * 512;
    k_pack<<<cdiv(ptot,256),256,0,stream>>>(Wm, Wp, Kreal, Nreal, Nt, ptot);
    int tiles = Mt * Nt;
    k_gemm<<<cdiv(tiles,8),256,0,stream>>>(Abf, lda, Wp, C, ldc, Mt, Nt, Kt);
  };

  // ---- gcn_norm ----
  Z(deg, N);
  k_deg <<<cdiv(E,256),256,0,stream>>>(dst, deg, E);
  k_dinv<<<cdiv(N,256),256,0,stream>>>(deg, N);
  k_norm<<<cdiv(E,256),256,0,stream>>>(src, dst, deg, norm, E);

  // ---- layer 1: propagate input-side (64-wide), out1 = sum_k (A^k x) W1[k] ----
  Z(out1, (long long)NP * H1);
  CVT(x, DIN, DIN, K1);                            GEMM(W1 + 0*DIN*H1, DIN, H1, K1/32, H1/16, K1, out1, H1);
  Z(hA, (long long)NP*DIN); PROP(x,  DIN, hA, DIN, DIN);
  CVT(hA, DIN, DIN, K1);                           GEMM(W1 + 1*DIN*H1, DIN, H1, K1/32, H1/16, K1, out1, H1);
  Z(hB, (long long)NP*DIN); PROP(hA, DIN, hB, DIN, DIN);
  CVT(hB, DIN, DIN, K1);                           GEMM(W1 + 2*DIN*H1, DIN, H1, K1/32, H1/16, K1, out1, H1);
  Z(hA, (long long)NP*DIN); PROP(hB, DIN, hA, DIN, DIN);
  CVT(hA, DIN, DIN, K1);                           GEMM(W1 + 3*DIN*H1, DIN, H1, K1/32, H1/16, K1, out1, H1);
  k_bias_relu<<<cdiv((long long)N*H1,256),256,0,stream>>>(out1, H1, b1, H1, (long long)N*H1);

  // ---- layer 2: Horner, propagate 200-wide: t = hW3; t = A t + hW2; ... ----
  CVT(out1, H1, H1, K2);   // h staged once, reused by all 4 GEMMs
  Z(gA, (long long)NP*LD2);                                  GEMM(W2 + 3*H1*H2, H1, H2, K2/32, LD2/16, K2, gA, LD2);
  Z(gB, (long long)NP*LD2); PROP(gA, LD2, gB, LD2, H2);      GEMM(W2 + 2*H1*H2, H1, H2, K2/32, LD2/16, K2, gB, LD2);
  Z(gA, (long long)NP*LD2); PROP(gB, LD2, gA, LD2, H2);      GEMM(W2 + 1*H1*H2, H1, H2, K2/32, LD2/16, K2, gA, LD2);
  Z(gB, (long long)NP*LD2); PROP(gA, LD2, gB, LD2, H2);      GEMM(W2 + 0*H1*H2, H1, H2, K2/32, LD2/16, K2, gB, LD2);
  k_bias_relu<<<cdiv((long long)N*H2,256),256,0,stream>>>(gB, LD2, b2, H2, (long long)N*H2);

  // ---- layer 3: Horner, propagate 4-wide; reuse hA/hB as (NP x 16) accumulators ----
  CVT(gB, LD2, H2, K3);
  Z(hA, (long long)NP*LD3);                                  GEMM(W3 + 3*H2*DOUT, H2, DOUT, K3/32, 1, K3, hA, LD3);
  Z(hB, (long long)NP*LD3); PROP(hA, LD3, hB, LD3, DOUT);    GEMM(W3 + 2*H2*DOUT, H2, DOUT, K3/32, 1, K3, hB, LD3);
  Z(hA, (long long)NP*LD3); PROP(hB, LD3, hA, LD3, DOUT);    GEMM(W3 + 1*H2*DOUT, H2, DOUT, K3/32, 1, K3, hA, LD3);
  Z(hB, (long long)NP*LD3); PROP(hA, LD3, hB, LD3, DOUT);    GEMM(W3 + 0*H2*DOUT, H2, DOUT, K3/32, 1, K3, hB, LD3);
  k_final<<<cdiv((long long)N*DOUT,256),256,0,stream>>>(hB, LD3, b3, out, N);
}
